// DynamicMemoryUpdater_63857573757527
// MI455X (gfx1250) — compile-verified, matrix operand order checked by reading, not run on hardware
//
#include <hip/hip_runtime.h>

// ---------------------------------------------------------------------------
// Types for CDNA5 WMMA (gfx1250, wave32)
// ---------------------------------------------------------------------------
typedef __attribute__((ext_vector_type(16))) __bf16 v16bf;
typedef __attribute__((ext_vector_type(8)))  float  v8f;

union BF16Frag {
    v16bf v;
    unsigned short u[16];
    uint4 q[2];
};

__device__ __forceinline__ unsigned short f2bf(float f) {
    unsigned int u = __float_as_uint(f);
    unsigned int r = u + 0x7FFFu + ((u >> 16) & 1u);   // round-to-nearest-even
    return (unsigned short)(r >> 16);
}
__device__ __forceinline__ float bf2f(unsigned short s) {
    return __uint_as_float(((unsigned int)s) << 16);
}

// gfx1250: generic pointer to LDS has the LDS byte offset in its low 32 bits
__device__ __forceinline__ unsigned lds_off(const void* p) {
    return (unsigned)(size_t)p;
}

// ---------------------------------------------------------------------------
// Zero scratch region (attended accumulators + counts) — required every call
// ---------------------------------------------------------------------------
__global__ void zero_u32_kernel(unsigned int* p, long n) {
    long i = (long)blockIdx.x * blockDim.x + threadIdx.x;
    if (i < n) p[i] = 0u;
}

// ---------------------------------------------------------------------------
// Transpose + cast weights: src f32 [R][C]  ->  dst bf16 [C][R]
// ---------------------------------------------------------------------------
__global__ void transpose_cast_kernel(const float* __restrict__ src,
                                      unsigned short* __restrict__ dst,
                                      int R, int C) {
    long i = (long)blockIdx.x * blockDim.x + threadIdx.x;
    long total = (long)R * C;
    if (i >= total) return;
    int c = (int)(i % C);
    int r = (int)(i / C);
    dst[(long)c * R + r] = f2bf(src[i]);
}

// ---------------------------------------------------------------------------
// Generic WMMA GEMM:  C[M][N] = act(A[M][K] @ Bt[N][K]^T + bias)
// A: f32 row-major (float4 loads, converted to bf16 while staging into LDS)
// Bt: bf16 row-major [N][K], staged via ASYNCcnt DMA global_load_async_to_lds
// Block: 256 threads = 8 waves, tile 128x128, K step 32.
// Wave (wm 0..3, wn 0..1) owns a 32x64 sub-tile = 2x4 wmma tiles.
// M % 128 == 0, N % 128 == 0, K % 32 == 0 at every call site.
// ---------------------------------------------------------------------------
#define GT_M 128
#define GT_N 128
#define GT_K 32
#define LDSPAD 8

__global__ __launch_bounds__(256) void gemm_bf16_wmma_kernel(
    const float* __restrict__ A, const unsigned short* __restrict__ Bt,
    const float* __restrict__ bias,
    float* __restrict__ outF, unsigned short* __restrict__ outBF,
    int Mdim, int Ndim, int Kdim, int relu)
{
    __shared__ unsigned short ldsA[GT_M][GT_K + LDSPAD];
    __shared__ unsigned short ldsB[GT_N][GT_K + LDSPAD];

    const int tid   = threadIdx.x;
    const int lane  = tid & 31;
    const int wave  = tid >> 5;
    const int waveM = wave & 3;      // 0..3 -> 32 rows each
    const int waveN = wave >> 2;     // 0..1 -> 64 cols each

    const int nBlocks = Ndim / GT_N;
    const int m0 = (blockIdx.x / nBlocks) * GT_M;
    const int n0 = (blockIdx.x % nBlocks) * GT_N;

    v8f acc[2][4];
    for (int i = 0; i < 2; ++i)
        for (int j = 0; j < 4; ++j)
            acc[i][j] = v8f{};

    const int lrow  = lane & 15;
    const int khalf = (lane >> 4) << 3;     // 0 or 8

    for (int k0 = 0; k0 < Kdim; k0 += GT_K) {
        // --- B tile (bf16): async DMA global -> LDS, 2 x b128 chunks/thread.
        //     Issued first so the DMA overlaps the VGPR-path A staging below.
        #pragma unroll
        for (int it = 0; it < 2; ++it) {
            int idx = tid + it * 256;          // 0..511 chunk id
            int r = idx >> 2;                  // 0..127
            int c = (idx & 3) << 3;            // ushort col: 0,8,16,24 (16B chunks)
            unsigned       loff = lds_off(&ldsB[r][c]);
            unsigned long long ga =
                (unsigned long long)(size_t)(Bt + (long)(n0 + r) * Kdim + k0 + c);
            asm volatile("global_load_async_to_lds_b128 %0, %1, off"
                         :: "v"(loff), "v"(ga) : "memory");
        }
        // --- A tile (f32 -> bf16): float4 load, pack 4 x bf16, b64 LDS store
        #pragma unroll
        for (int it = 0; it < 4; ++it) {
            int idx = tid + it * 256;          // 0..1023
            int r = idx >> 3;                  // 0..127
            int c = (idx & 7) << 2;            // float col: 0,4,...,28
            const float* ap = A + (long)(m0 + r) * Kdim + k0 + c;
            float4 f = *(const float4*)ap;
            if (k0 + GT_K < Kdim) __builtin_prefetch(ap + GT_K, 0, 1);
            unsigned lo = (unsigned)f2bf(f.x) | ((unsigned)f2bf(f.y) << 16);
            unsigned hi = (unsigned)f2bf(f.z) | ((unsigned)f2bf(f.w) << 16);
            *(unsigned long long*)&ldsA[r][c] =
                (unsigned long long)lo | ((unsigned long long)hi << 32);
        }
        asm volatile("s_wait_asynccnt 0" ::: "memory");
        __syncthreads();

        BF16Frag afr[2], bfr[4];
        for (int ti = 0; ti < 2; ++ti) {
            int row = waveM * 32 + ti * 16 + lrow;
            #pragma unroll
            for (int i = 0; i < 16; ++i) {
                int kk = (i < 8 ? i : i + 8) + khalf;
                afr[ti].u[i] = ldsA[row][kk];
            }
        }
        for (int tj = 0; tj < 4; ++tj) {
            int col = waveN * 64 + tj * 16 + lrow;
            #pragma unroll
            for (int i = 0; i < 16; ++i) {
                int kk = (i < 8 ? i : i + 8) + khalf;
                bfr[tj].u[i] = ldsB[col][kk];
            }
        }
        #pragma unroll
        for (int ti = 0; ti < 2; ++ti)
            #pragma unroll
            for (int tj = 0; tj < 4; ++tj)
                acc[ti][tj] = __builtin_amdgcn_wmma_f32_16x16x32_bf16(
                    false, afr[ti].v, false, bfr[tj].v,
                    (short)0, acc[ti][tj], false, false);
        __syncthreads();
    }

    // Write back.  C layout: vgpr r -> M = r + 8*(lane>=16), N = lane&15
    const int mrOff = (lane >> 4) << 3;
    for (int ti = 0; ti < 2; ++ti) {
        for (int tj = 0; tj < 4; ++tj) {
            int nn = n0 + waveN * 64 + tj * 16 + lrow;
            float bv = bias ? bias[nn] : 0.0f;
            #pragma unroll
            for (int r = 0; r < 8; ++r) {
                int mm = m0 + waveM * 32 + ti * 16 + mrOff + r;
                float v = acc[ti][tj][r] + bv;
                if (relu) v = fmaxf(v, 0.0f);
                long o = (long)mm * Ndim + nn;
                if (outF)  outF[o]  = v;
                if (outBF) outBF[o] = f2bf(v);
            }
        }
    }
}

// ---------------------------------------------------------------------------
// Fused scores + top-k + scatter kernel.
// grid = (Ntok/32, H).  Block 256 threads = 8 waves; wave w computes score
// rows m in [w*32, w*32+32) for all 32 tokens of this block -> the block
// covers the FULL 256-slot axis, so top-k can be done in LDS.
// ---------------------------------------------------------------------------
__global__ __launch_bounds__(256) void scores_topk_kernel(
    const unsigned short* __restrict__ qsbf,   // [256][1024] bf16
    const unsigned short* __restrict__ kvbf,   // [Ntok][1024] bf16
    const float* __restrict__ memory_bias,     // [8][256]
    float* __restrict__ out,                   // d_out base (f32)
    float* __restrict__ attended,              // [8][256][128]
    int*   __restrict__ counts,                // [8][256]
    long scOff, long gateOff, int Ntok)
{
    __shared__ float ldsS[256][33];
    __shared__ int   topIdx[32][4];

    const int h     = blockIdx.y;
    const int nBase = blockIdx.x * 32;
    const int tid   = threadIdx.x;
    const int lane  = tid & 31;
    const int wave  = tid >> 5;
    const int lrow  = lane & 15;
    const int khalf = (lane >> 4) << 3;

    v8f acc[2][2];
    for (int i = 0; i < 2; ++i)
        for (int j = 0; j < 2; ++j)
            acc[i][j] = v8f{};

    for (int k0 = 0; k0 < 128; k0 += 32) {
        BF16Frag afr[2], bfr[2];
        #pragma unroll
        for (int ti = 0; ti < 2; ++ti) {
            int m = wave * 32 + ti * 16 + lrow;
            const uint4* p = (const uint4*)(qsbf + (long)m * 1024 + h * 128 + k0 + khalf);
            afr[ti].q[0] = p[0];
            afr[ti].q[1] = p[2];     // +16 halfwords
        }
        #pragma unroll
        for (int tj = 0; tj < 2; ++tj) {
            int n = nBase + tj * 16 + lrow;
            const uint4* p = (const uint4*)(kvbf + (long)n * 1024 + h * 128 + k0 + khalf);
            bfr[tj].q[0] = p[0];
            bfr[tj].q[1] = p[2];
        }
        #pragma unroll
        for (int ti = 0; ti < 2; ++ti)
            #pragma unroll
            for (int tj = 0; tj < 2; ++tj)
                acc[ti][tj] = __builtin_amdgcn_wmma_f32_16x16x32_bf16(
                    false, afr[ti].v, false, bfr[tj].v,
                    (short)0, acc[ti][tj], false, false);
    }

    // Scale, add 5*memory_bias, stream (non-temporal) to both output regions,
    // stash in LDS for the top-k pass.
    const float scale = 0.08838834764831845f;   // 1/sqrt(128)
    const int mrOff = (lane >> 4) << 3;
    for (int ti = 0; ti < 2; ++ti) {
        for (int tj = 0; tj < 2; ++tj) {
            #pragma unroll
            for (int r = 0; r < 8; ++r) {
                int m  = wave * 32 + ti * 16 + mrOff + r;
                int nl = tj * 16 + lrow;
                float v = acc[ti][tj][r] * scale + 5.0f * memory_bias[h * 256 + m];
                long o = (long)(h * 256 + m) * Ntok + nBase + nl;
                __builtin_nontemporal_store(v, &out[scOff + o]);
                __builtin_nontemporal_store(v, &out[gateOff + o]);
                ldsS[m][nl] = v;
            }
        }
    }
    __syncthreads();

    // Top-4 over the 256 slots, one column (token) per thread 0..31.
    if (tid < 32) {
        float bv0 = -3.4e38f, bv1 = -3.4e38f, bv2 = -3.4e38f, bv3 = -3.4e38f;
        int   bi0 = 0, bi1 = 0, bi2 = 0, bi3 = 0;
        for (int m = 0; m < 256; ++m) {
            float v = ldsS[m][tid];
            if (v > bv3) {
                if (v > bv0)      { bv3=bv2;bi3=bi2; bv2=bv1;bi2=bi1; bv1=bv0;bi1=bi0; bv0=v;bi0=m; }
                else if (v > bv1) { bv3=bv2;bi3=bi2; bv2=bv1;bi2=bi1; bv1=v;bi1=m; }
                else if (v > bv2) { bv3=bv2;bi3=bi2; bv2=v;bi2=m; }
                else              { bv3=v;bi3=m; }
            }
        }
        topIdx[tid][0] = bi0; topIdx[tid][1] = bi1;
        topIdx[tid][2] = bi2; topIdx[tid][3] = bi3;
    }
    __syncthreads();

    // Scatter-add: attended[h][slot][:] += k[h][n][:]; counts[h][slot] += 1.
    // 128 (token, rank) pairs, 2 threads per pair, 64 dims each.
    int pair = tid >> 1;
    int half = tid & 1;
    int col  = pair >> 2;
    int rank = pair & 3;
    int slot = topIdx[col][rank];
    int n    = nBase + col;
    const unsigned short* kp = kvbf + (long)n * 1024 + h * 128;
    float* ap = attended + (long)(h * 256 + slot) * 128;
    for (int d = half * 64; d < half * 64 + 64; ++d)
        atomicAdd(&ap[d], bf2f(kp[d]));
    if (half == 0)
        atomicAdd(&counts[h * 256 + slot], 1);
}

// ---------------------------------------------------------------------------
// concat([mem, context]) -> LayerNorm -> h1 (f32 [256][2048])
// One block per slot row, 256 threads, 8 elements per thread.
// ---------------------------------------------------------------------------
__global__ __launch_bounds__(256) void concat_ln_kernel(
    const float* __restrict__ mem, const float* __restrict__ attended,
    const float* __restrict__ g, const float* __restrict__ b,
    float* __restrict__ h1)
{
    __shared__ float red[256];
    const int m = blockIdx.x, tid = threadIdx.x;
    float x[8];
    float s = 0.0f;
    #pragma unroll
    for (int i = 0; i < 8; ++i) {
        int j = tid + i * 256;
        float v;
        if (j < 1024) v = mem[(long)m * 1024 + j];
        else {
            int c = j - 1024;
            v = attended[((long)(c >> 7) * 256 + m) * 128 + (c & 127)];
        }
        x[i] = v; s += v;
    }
    red[tid] = s; __syncthreads();
    for (int off = 128; off > 0; off >>= 1) {
        if (tid < off) red[tid] += red[tid + off];
        __syncthreads();
    }
    float mu = red[0] / 2048.0f;
    __syncthreads();
    float s2 = 0.0f;
    #pragma unroll
    for (int i = 0; i < 8; ++i) { float d = x[i] - mu; s2 += d * d; }
    red[tid] = s2; __syncthreads();
    for (int off = 128; off > 0; off >>= 1) {
        if (tid < off) red[tid] += red[tid + off];
        __syncthreads();
    }
    float inv = rsqrtf(red[0] / 2048.0f + 1e-5f);
    #pragma unroll
    for (int i = 0; i < 8; ++i) {
        int j = tid + i * 256;
        h1[(long)m * 2048 + j] = (x[i] - mu) * inv * g[j] + b[j];
    }
}

// ---------------------------------------------------------------------------
// Output LayerNorm: h2 [256][1024] -> dyn (d_out[0..262143])
// ---------------------------------------------------------------------------
__global__ __launch_bounds__(256) void ln_out_kernel(
    const float* __restrict__ h2, const float* __restrict__ g,
    const float* __restrict__ b, float* __restrict__ outDyn)
{
    __shared__ float red[256];
    const int m = blockIdx.x, tid = threadIdx.x;
    float x[4];
    float s = 0.0f;
    #pragma unroll
    for (int i = 0; i < 4; ++i) {
        x[i] = h2[(long)m * 1024 + tid + i * 256];
        s += x[i];
    }
    red[tid] = s; __syncthreads();
    for (int off = 128; off > 0; off >>= 1) {
        if (tid < off) red[tid] += red[tid + off];
        __syncthreads();
    }
    float mu = red[0] / 1024.0f;
    __syncthreads();
    float s2 = 0.0f;
    #pragma unroll
    for (int i = 0; i < 4; ++i) { float d = x[i] - mu; s2 += d * d; }
    red[tid] = s2; __syncthreads();
    for (int off = 128; off > 0; off >>= 1) {
        if (tid < off) red[tid] += red[tid + off];
        __syncthreads();
    }
    float inv = rsqrtf(red[0] / 1024.0f + 1e-5f);
    #pragma unroll
    for (int i = 0; i < 4; ++i) {
        int j = tid + i * 256;
        outDyn[(long)m * 1024 + j] = (x[i] - mu) * inv * g[j] + b[j];
    }
}

// ---------------------------------------------------------------------------
// load_fraction[m] = mean over heads of counts[h][m]
// ---------------------------------------------------------------------------
__global__ void load_fraction_kernel(const int* __restrict__ counts,
                                     float* __restrict__ outLF) {
    int m = threadIdx.x;
    int s = 0;
    for (int h = 0; h < 8; ++h) s += counts[h * 256 + m];
    outLF[m] = (float)s * 0.125f;
}

// ---------------------------------------------------------------------------
// Host-side launch
// ---------------------------------------------------------------------------
extern "C" void kernel_launch(void* const* d_in, const int* in_sizes, int n_in,
                              void* d_out, int out_size, void* d_ws, size_t ws_size,
                              hipStream_t stream) {
    const float* flat = (const float*)d_in[0];   // [32768][2048]
    const float* mem  = (const float*)d_in[1];   // [256][1024]
    const float* Wq   = (const float*)d_in[2];
    const float* bq   = (const float*)d_in[3];
    const float* Ws1  = (const float*)d_in[4];
    const float* bs1  = (const float*)d_in[5];
    const float* Ws2  = (const float*)d_in[6];
    const float* bs2  = (const float*)d_in[7];
    const float* mbias= (const float*)d_in[8];   // [8][256]
    const float* ln1g = (const float*)d_in[9];
    const float* ln1b = (const float*)d_in[10];
    const float* Wu1  = (const float*)d_in[11];
    const float* bu1  = (const float*)d_in[12];
    const float* Wu2  = (const float*)d_in[13];
    const float* bu2  = (const float*)d_in[14];
    const float* lnog = (const float*)d_in[15];
    const float* lnob = (const float*)d_in[16];

    const int NTOK = 32768;
    float* out = (float*)d_out;
    const long DYN   = 256L * 1024;
    const long SC    = 8L * 256 * NTOK;
    const long scOff = DYN;
    const long gateOff = DYN + SC;
    const long lfOff = DYN + 2 * SC;

    // Workspace carve-up (256B aligned slices)
    char* ws = (char*)d_ws;
    size_t off = 0;
    auto alloc = [&](size_t bytes) {
        size_t cur = off;
        off += (bytes + 255) & ~(size_t)255;
        return cur;
    };
    unsigned short* WqT  = (unsigned short*)(ws + alloc(1024L * 2048 * 2));
    unsigned short* Ws1T = (unsigned short*)(ws + alloc(4096L * 1024 * 2));
    unsigned short* Ws2T = (unsigned short*)(ws + alloc(1024L * 4096 * 2));
    unsigned short* Wu1T = (unsigned short*)(ws + alloc(2048L * 2048 * 2));
    unsigned short* Wu2T = (unsigned short*)(ws + alloc(1024L * 2048 * 2));
    unsigned short* kvbf = (unsigned short*)(ws + alloc((long)NTOK * 1024 * 2));
    float*          t1   = (float*)(ws + alloc(256L * 4096 * 4));
    unsigned short* qsbf = (unsigned short*)(ws + alloc(256L * 1024 * 2));
    float*       attended= (float*)(ws + alloc(8L * 256 * 128 * 4));   // contiguous with
    int*         counts  = (int*)  (ws + alloc(2048L * 4));            // counts (zeroed together)
    float*          h1   = (float*)(ws + alloc(256L * 2048 * 4));
    float*          hh   = (float*)(ws + alloc(256L * 2048 * 4));
    float*          h2   = (float*)(ws + alloc(256L * 1024 * 4));

    // 1. zero accumulators (attended + counts are contiguous)
    {
        long n = 8L * 256 * 128 + 2048;
        zero_u32_kernel<<<(int)((n + 255) / 256), 256, 0, stream>>>((unsigned int*)attended, n);
    }
    // 2. weight transpose+cast to bf16
    {
        long n;
        n = 2048L * 1024; transpose_cast_kernel<<<(int)((n+255)/256),256,0,stream>>>(Wq,  WqT,  2048, 1024);
        n = 1024L * 4096; transpose_cast_kernel<<<(int)((n+255)/256),256,0,stream>>>(Ws1, Ws1T, 1024, 4096);
        n = 4096L * 1024; transpose_cast_kernel<<<(int)((n+255)/256),256,0,stream>>>(Ws2, Ws2T, 4096, 1024);
        n = 2048L * 2048; transpose_cast_kernel<<<(int)((n+255)/256),256,0,stream>>>(Wu1, Wu1T, 2048, 2048);
        n = 2048L * 1024; transpose_cast_kernel<<<(int)((n+255)/256),256,0,stream>>>(Wu2, Wu2T, 2048, 1024);
    }
    // 3. kv = flat @ Wq + bq  (bf16 out only)
    gemm_bf16_wmma_kernel<<<(NTOK/128)*(1024/128), 256, 0, stream>>>(
        flat, WqT, bq, nullptr, kvbf, NTOK, 1024, 2048, 0);
    // 4. t1 = relu(mem @ Ws1 + bs1)
    gemm_bf16_wmma_kernel<<<(256/128)*(4096/128), 256, 0, stream>>>(
        mem, Ws1T, bs1, t1, nullptr, 256, 4096, 1024, 1);
    // 5. q_slots = t1 @ Ws2 + bs2 (bf16 out)
    gemm_bf16_wmma_kernel<<<(256/128)*(1024/128), 256, 0, stream>>>(
        t1, Ws2T, bs2, nullptr, qsbf, 256, 1024, 4096, 0);
    // 6. fused scores + bias + top-4 + scatter (writes both score output regions)
    {
        dim3 grid(NTOK / 32, 8);
        scores_topk_kernel<<<grid, 256, 0, stream>>>(
            qsbf, kvbf, mbias, out, attended, counts, scOff, gateOff, NTOK);
    }
    // 7. concat + LN -> h1
    concat_ln_kernel<<<256, 256, 0, stream>>>(mem, attended, ln1g, ln1b, h1);
    // 8. hh = relu(h1 @ Wu1 + bu1)
    gemm_bf16_wmma_kernel<<<(256/128)*(2048/128), 256, 0, stream>>>(
        h1, Wu1T, bu1, hh, nullptr, 256, 2048, 2048, 1);
    // 9. h2 = hh @ Wu2 + bu2
    gemm_bf16_wmma_kernel<<<(256/128)*(1024/128), 256, 0, stream>>>(
        hh, Wu2T, bu2, h2, nullptr, 256, 1024, 2048, 0);
    // 10. dyn = LN(h2) -> d_out[0:262144]
    ln_out_kernel<<<256, 256, 0, stream>>>(h2, lnog, lnob, out);
    // 11. load_fraction
    load_fraction_kernel<<<1, 256, 0, stream>>>(counts, out + lfOff);
}
